// Mamba_20899310862721
// MI455X (gfx1250) — matrix operand order repaired and checked by reading, not compiled, not run on hardware
//
#include <hip/hip_runtime.h>

typedef __bf16 bf16;
typedef __attribute__((ext_vector_type(16))) __bf16       v16bf;
typedef __attribute__((ext_vector_type(8)))  float        v8f;
typedef __attribute__((ext_vector_type(4)))  unsigned int u32x4;

#define SEQ_LEN 2048
#define VOCAB   32000
#define N_EMBD  768
#define D_INNER 1536
#define D_STATE 16
#define D_CONV  4
#define DT_RANK 48
#define XDBL_W  (DT_RANK + 2 * D_STATE)   // 80 logical
#define XDBL_P  128                        // padded row stride for x_proj output
#define DT_KP   64                         // dt_proj K padded 48 -> 64

union FragAB { v16bf v; u32x4 q[2]; };
union FragC  { v8f   v; float f[8]; };

__device__ __forceinline__ float siluf(float x) { return x / (1.0f + __expf(-x)); }

__device__ __forceinline__ void load_a(FragAB& a, const bf16* Arow, int k0, int half)
{
    const bf16* Ap = Arow + k0 + 8 * half;
    a.q[0] = *(const u32x4*)(Ap);        // K = k0+8h   .. +7
    a.q[1] = *(const u32x4*)(Ap + 16);   // K = k0+16+8h .. +7
}

__device__ __forceinline__ void load_b(FragAB& b, const bf16* Wrow, int k0, int half)
{
    const bf16* Bp = Wrow + k0 + 8 * half;
    b.q[0] = *(const u32x4*)(Bp);
    b.q[1] = *(const u32x4*)(Bp + 16);
}

// ---------------------------------------------------------------------------
// bf16 WMMA GEMM, guard-free, 2-stage software pipeline.
// C[M,N] = A[M,K] * W[N,K]^T  (f32 accumulate)
// Requires: M % 64 == 0 (grid.y = M/64), N % 64 == 0 (grid.x = N/64), K % 64 == 0.
// epi: 0 = store, 1 = softplus(C + bias[n]), 2 = C[m,n] += acc (residual)
// Block = 128 threads (4 wave32); each wave computes a 16x64 strip.
// ---------------------------------------------------------------------------
__global__ __launch_bounds__(128)
void wmma_gemm_bf16(const bf16* __restrict__ A, int lda,
                    const bf16* __restrict__ W, int ldw,
                    float* __restrict__ C, int ldc,
                    int K, int epi, const float* __restrict__ bias)
{
    const int wave = threadIdx.x >> 5;
    const int lane = threadIdx.x & 31;
    const int half = lane >> 4;
    const int l16  = lane & 15;
    const int m0 = blockIdx.y * 64 + wave * 16;
    const int n0 = blockIdx.x * 64;

    FragC acc[4];
    #pragma unroll
    for (int i = 0; i < 4; ++i)
        #pragma unroll
        for (int e = 0; e < 8; ++e) acc[i].f[e] = 0.0f;

    const bf16* Arow  = A + (size_t)(m0 + l16) * lda;
    const bf16* Wrow0 = W + (size_t)(n0 +  0 + l16) * ldw;
    const bf16* Wrow1 = W + (size_t)(n0 + 16 + l16) * ldw;
    const bf16* Wrow2 = W + (size_t)(n0 + 32 + l16) * ldw;
    const bf16* Wrow3 = W + (size_t)(n0 + 48 + l16) * ldw;

    // Stage 0 fragments (K-chunk k0) and stage 1 fragments (k0+32)
    FragAB a0, p0, p1, p2, p3;   // current
    FragAB a1, q0, q1, q2, q3;   // next

    load_a(a0, Arow, 0, half);
    load_b(p0, Wrow0, 0, half);
    load_b(p1, Wrow1, 0, half);
    load_b(p2, Wrow2, 0, half);
    load_b(p3, Wrow3, 0, half);

    for (int k0 = 0; k0 < K; k0 += 64) {
        // issue loads for k0+32 (always in range since K % 64 == 0)
        load_a(a1, Arow, k0 + 32, half);
        load_b(q0, Wrow0, k0 + 32, half);
        load_b(q1, Wrow1, k0 + 32, half);
        load_b(q2, Wrow2, k0 + 32, half);
        load_b(q3, Wrow3, k0 + 32, half);

        acc[0].v = __builtin_amdgcn_wmma_f32_16x16x32_bf16(
            false, a0.v, false, p0.v, (short)0, acc[0].v, false, false);
        acc[1].v = __builtin_amdgcn_wmma_f32_16x16x32_bf16(
            false, a0.v, false, p1.v, (short)0, acc[1].v, false, false);
        acc[2].v = __builtin_amdgcn_wmma_f32_16x16x32_bf16(
            false, a0.v, false, p2.v, (short)0, acc[2].v, false, false);
        acc[3].v = __builtin_amdgcn_wmma_f32_16x16x32_bf16(
            false, a0.v, false, p3.v, (short)0, acc[3].v, false, false);

        if (k0 + 64 < K) {
            // issue loads for k0+64 while stage-1 WMMAs run
            load_a(a0, Arow, k0 + 64, half);
            load_b(p0, Wrow0, k0 + 64, half);
            load_b(p1, Wrow1, k0 + 64, half);
            load_b(p2, Wrow2, k0 + 64, half);
            load_b(p3, Wrow3, k0 + 64, half);
            __builtin_prefetch(Arow  + k0 + 128, 0, 3);
            __builtin_prefetch(Wrow0 + k0 + 128, 0, 3);
        }

        acc[0].v = __builtin_amdgcn_wmma_f32_16x16x32_bf16(
            false, a1.v, false, q0.v, (short)0, acc[0].v, false, false);
        acc[1].v = __builtin_amdgcn_wmma_f32_16x16x32_bf16(
            false, a1.v, false, q1.v, (short)0, acc[1].v, false, false);
        acc[2].v = __builtin_amdgcn_wmma_f32_16x16x32_bf16(
            false, a1.v, false, q2.v, (short)0, acc[2].v, false, false);
        acc[3].v = __builtin_amdgcn_wmma_f32_16x16x32_bf16(
            false, a1.v, false, q3.v, (short)0, acc[3].v, false, false);
    }

    // Epilogue: C/D layout — VGPR j holds row m0 + j + 8*half, col n0 + nt*16 + l16
    #pragma unroll
    for (int nt = 0; nt < 4; ++nt) {
        int n = n0 + nt * 16 + l16;
        #pragma unroll
        for (int j = 0; j < 8; ++j) {
            int m = m0 + j + 8 * half;
            float v = acc[nt].f[j];
            size_t idx = (size_t)m * ldc + n;
            if (epi == 1) {
                v += bias[n];
                C[idx] = (v > 20.0f) ? v : log1pf(__expf(v));
            } else if (epi == 2) {
                C[idx] += v;
            } else {
                C[idx] = v;
            }
        }
    }
}

// ---------------------------------------------------------------------------
__global__ void f32_to_bf16_kernel(const float* __restrict__ in,
                                   bf16* __restrict__ out, int n)
{
    int i = blockIdx.x * blockDim.x + threadIdx.x;
    if (i < n) out[i] = (bf16)in[i];
}

__global__ void zero_bf16_kernel(bf16* __restrict__ p, int n)
{
    int i = blockIdx.x * blockDim.x + threadIdx.x;
    if (i < n) p[i] = (bf16)0.0f;
}

// pack xdbl[:, :48] (row stride XDBL_P) into [SEQ_LEN, 64] bf16, zero tail cols
__global__ void pack_dtA_kernel(const float* __restrict__ xdbl,
                                bf16* __restrict__ out)
{
    int i = blockIdx.x * blockDim.x + threadIdx.x;
    if (i >= SEQ_LEN * DT_KP) return;
    int t = i / DT_KP, c = i % DT_KP;
    out[i] = (c < DT_RANK) ? (bf16)xdbl[(size_t)t * XDBL_P + c] : (bf16)0.0f;
}

// pack dt_proj_w [D_INNER, 48] into [D_INNER, 64] bf16, zero tail cols
__global__ void pack_dtW_kernel(const float* __restrict__ w,
                                bf16* __restrict__ out)
{
    int i = blockIdx.x * blockDim.x + threadIdx.x;
    if (i >= D_INNER * DT_KP) return;
    int r = i / DT_KP, c = i % DT_KP;
    out[i] = (c < DT_RANK) ? (bf16)w[(size_t)r * DT_RANK + c] : (bf16)0.0f;
}

__global__ void gather_kernel(const int* __restrict__ tokens,
                              const float* __restrict__ E,
                              float* __restrict__ h)
{
    int t = blockIdx.x;
    int tok = tokens[t];
    for (int i = threadIdx.x; i < N_EMBD; i += blockDim.x)
        h[(size_t)t * N_EMBD + i] = E[(size_t)tok * N_EMBD + i];
}

__global__ __launch_bounds__(256)
void rmsnorm_kernel(const float* __restrict__ x, const float* __restrict__ w,
                    const float* __restrict__ b, float* __restrict__ out,
                    bf16* __restrict__ out_bf)
{
    __shared__ float red[256];
    const int t = blockIdx.x;
    const float* row = x + (size_t)t * N_EMBD;
    float v[3], s = 0.0f;
    #pragma unroll
    for (int i = 0; i < 3; ++i) { v[i] = row[threadIdx.x + 256 * i]; s += v[i] * v[i]; }
    red[threadIdx.x] = s;
    __syncthreads();
    for (int off = 128; off > 0; off >>= 1) {
        if (threadIdx.x < off) red[threadIdx.x] += red[threadIdx.x + off];
        __syncthreads();
    }
    float inv = rsqrtf(red[0] / (float)N_EMBD + 1e-5f);
    #pragma unroll
    for (int i = 0; i < 3; ++i) {
        int c = threadIdx.x + 256 * i;
        float o = v[i] * inv * w[c] + b[c];
        out[(size_t)t * N_EMBD + c]    = o;
        out_bf[(size_t)t * N_EMBD + c] = (bf16)o;
    }
}

// depthwise causal conv (taps=4) + bias + SiLU; input is x half of x_and_res
__global__ void conv_silu_kernel(const float* __restrict__ xr,
                                 const float* __restrict__ cw,
                                 const float* __restrict__ cb,
                                 float* __restrict__ xc, bf16* __restrict__ xc_bf)
{
    int idx = blockIdx.x * blockDim.x + threadIdx.x;
    if (idx >= SEQ_LEN * D_INNER) return;
    int t = idx / D_INNER, d = idx % D_INNER;
    float s = cb[d];
    #pragma unroll
    for (int j = 0; j < D_CONV; ++j) {
        int tt = t - (D_CONV - 1) + j;
        if (tt >= 0) s += cw[d * D_CONV + j] * xr[(size_t)tt * (2 * D_INNER) + d];
    }
    s = siluf(s);
    xc[idx] = s;
    xc_bf[idx] = (bf16)s;
}

// selective scan: sequential over L, parallel over the 1536 channels
__global__ __launch_bounds__(256)
void scan_kernel(const float* __restrict__ xc, const float* __restrict__ delta,
                 const float* __restrict__ xdbl, const float* __restrict__ A_log,
                 const float* __restrict__ Dp, float* __restrict__ y)
{
    __shared__ float sB[D_STATE], sC[D_STATE];
    int d = blockIdx.x * blockDim.x + threadIdx.x;   // 6 blocks * 256 = 1536
    float A[D_STATE], st[D_STATE];
    #pragma unroll
    for (int n = 0; n < D_STATE; ++n) {
        A[n] = -__expf(A_log[(size_t)d * D_STATE + n]);
        st[n] = 0.0f;
    }
    float Dd = Dp[d];
    for (int t = 0; t < SEQ_LEN; ++t) {
        __syncthreads();
        if (threadIdx.x < 2 * D_STATE) {
            float v = xdbl[(size_t)t * XDBL_P + DT_RANK + threadIdx.x];
            if (threadIdx.x < D_STATE) sB[threadIdx.x] = v;
            else                       sC[threadIdx.x - D_STATE] = v;
        }
        __syncthreads();
        float dt = delta[(size_t)t * D_INNER + d];
        float u  = xc[(size_t)t * D_INNER + d];
        float du = dt * u, acc = 0.0f;
        #pragma unroll
        for (int n = 0; n < D_STATE; ++n) {
            st[n] = __expf(dt * A[n]) * st[n] + du * sB[n];
            acc += st[n] * sC[n];
        }
        y[(size_t)t * D_INNER + d] = acc + u * Dd;
    }
}

// y *= silu(res); also emit bf16 copy for out_proj GEMM
__global__ void gate_kernel(float* __restrict__ y, const float* __restrict__ xr,
                            bf16* __restrict__ y_bf)
{
    int idx = blockIdx.x * blockDim.x + threadIdx.x;
    if (idx >= SEQ_LEN * D_INNER) return;
    int t = idx / D_INNER, d = idx % D_INNER;
    float r = xr[(size_t)t * (2 * D_INNER) + D_INNER + d];
    float v = y[idx] * siluf(r);
    y[idx] = v;
    y_bf[idx] = (bf16)v;
}

// ---------------------------------------------------------------------------
extern "C" void kernel_launch(void* const* d_in, const int* in_sizes, int n_in,
                              void* d_out, int out_size, void* d_ws, size_t ws_size,
                              hipStream_t stream)
{
    const int*   tokens = (const int*)  d_in[0];
    const float* E      = (const float*)d_in[1];
    const float* in_w   = (const float*)d_in[2];
    const float* cw     = (const float*)d_in[3];
    const float* cb     = (const float*)d_in[4];
    const float* xp_w   = (const float*)d_in[5];
    const float* dt_w   = (const float*)d_in[6];
    const float* dt_b   = (const float*)d_in[7];
    const float* A_log  = (const float*)d_in[8];
    const float* Dp     = (const float*)d_in[9];
    const float* out_w  = (const float*)d_in[10];
    const float* nw     = (const float*)d_in[11];
    const float* nb     = (const float*)d_in[12];
    const float* nfw    = (const float*)d_in[13];
    const float* nfb    = (const float*)d_in[14];
    (void)in_sizes; (void)n_in; (void)out_size; (void)ws_size;

    char* ws = (char*)d_ws;
    size_t off = 0;
    auto alloc = [&](size_t bytes) -> char* {
        char* p = ws + off;
        off = (off + bytes + 255) & ~(size_t)255;
        return p;
    };
    bf16*  E_bf   = (bf16*) alloc((size_t)VOCAB * N_EMBD * 2);
    float* h      = (float*)alloc((size_t)SEQ_LEN * N_EMBD * 4);
    float* xn     = (float*)alloc((size_t)SEQ_LEN * N_EMBD * 4);
    bf16*  xn_bf  = (bf16*) alloc((size_t)SEQ_LEN * N_EMBD * 2);
    float* xr     = (float*)alloc((size_t)SEQ_LEN * 2 * D_INNER * 4);
    float* xc     = (float*)alloc((size_t)SEQ_LEN * D_INNER * 4);
    bf16*  xc_bf  = (bf16*) alloc((size_t)SEQ_LEN * D_INNER * 2);
    float* xdbl   = (float*)alloc((size_t)SEQ_LEN * XDBL_P * 4);     // padded stride
    bf16*  dtA_bf = (bf16*) alloc((size_t)SEQ_LEN * DT_KP * 2);
    bf16*  dtW_bf = (bf16*) alloc((size_t)D_INNER * DT_KP * 2);
    float* dl     = (float*)alloc((size_t)SEQ_LEN * D_INNER * 4);
    float* yb     = (float*)alloc((size_t)SEQ_LEN * D_INNER * 4);
    bf16*  y_bf   = (bf16*) alloc((size_t)SEQ_LEN * D_INNER * 2);
    bf16*  Wbf    = (bf16*) alloc((size_t)2 * D_INNER * N_EMBD * 2); // max weight

    const dim3 blk(256);
    const dim3 gblk(128);

    f32_to_bf16_kernel<<<(VOCAB * N_EMBD + 255) / 256, blk, 0, stream>>>(
        E, E_bf, VOCAB * N_EMBD);
    gather_kernel<<<SEQ_LEN, blk, 0, stream>>>(tokens, E, h);

    for (int l = 0; l < 2; ++l) {
        rmsnorm_kernel<<<SEQ_LEN, blk, 0, stream>>>(
            h, nw + l * N_EMBD, nb + l * N_EMBD, xn, xn_bf);

        // in_proj: [2048,768] x [3072,768]^T -> [2048,3072]
        f32_to_bf16_kernel<<<(2 * D_INNER * N_EMBD + 255) / 256, blk, 0, stream>>>(
            in_w + (size_t)l * 2 * D_INNER * N_EMBD, Wbf, 2 * D_INNER * N_EMBD);
        wmma_gemm_bf16<<<dim3((2 * D_INNER) / 64, SEQ_LEN / 64), gblk, 0, stream>>>(
            xn_bf, N_EMBD, Wbf, N_EMBD, xr, 2 * D_INNER, N_EMBD, 0, nullptr);

        conv_silu_kernel<<<(SEQ_LEN * D_INNER + 255) / 256, blk, 0, stream>>>(
            xr, cw + (size_t)l * D_INNER * D_CONV, cb + (size_t)l * D_INNER, xc, xc_bf);

        // x_proj: [2048,1536] x [80,1536]^T -> [2048,80], padded N to 128
        f32_to_bf16_kernel<<<(XDBL_W * D_INNER + 255) / 256, blk, 0, stream>>>(
            xp_w + (size_t)l * XDBL_W * D_INNER, Wbf, XDBL_W * D_INNER);
        zero_bf16_kernel<<<((XDBL_P - XDBL_W) * D_INNER + 255) / 256, blk, 0, stream>>>(
            Wbf + (size_t)XDBL_W * D_INNER, (XDBL_P - XDBL_W) * D_INNER);
        wmma_gemm_bf16<<<dim3(XDBL_P / 64, SEQ_LEN / 64), gblk, 0, stream>>>(
            xc_bf, D_INNER, Wbf, D_INNER, xdbl, XDBL_P, D_INNER, 0, nullptr);

        // dt_proj: pack K 48 -> 64 (zero tail), then [2048,64] x [1536,64]^T
        pack_dtA_kernel<<<(SEQ_LEN * DT_KP + 255) / 256, blk, 0, stream>>>(xdbl, dtA_bf);
        pack_dtW_kernel<<<(D_INNER * DT_KP + 255) / 256, blk, 0, stream>>>(
            dt_w + (size_t)l * D_INNER * DT_RANK, dtW_bf);
        wmma_gemm_bf16<<<dim3(D_INNER / 64, SEQ_LEN / 64), gblk, 0, stream>>>(
            dtA_bf, DT_KP, dtW_bf, DT_KP, dl, D_INNER, DT_KP, 1,
            dt_b + (size_t)l * D_INNER);

        scan_kernel<<<D_INNER / 256, blk, 0, stream>>>(
            xc, dl, xdbl, A_log + (size_t)l * D_INNER * D_STATE,
            Dp + (size_t)l * D_INNER, yb);

        gate_kernel<<<(SEQ_LEN * D_INNER + 255) / 256, blk, 0, stream>>>(yb, xr, y_bf);

        // out_proj: [2048,1536] x [768,1536]^T, accumulate into residual h
        f32_to_bf16_kernel<<<(N_EMBD * D_INNER + 255) / 256, blk, 0, stream>>>(
            out_w + (size_t)l * N_EMBD * D_INNER, Wbf, N_EMBD * D_INNER);
        wmma_gemm_bf16<<<dim3(N_EMBD / 64, SEQ_LEN / 64), gblk, 0, stream>>>(
            y_bf, D_INNER, Wbf, D_INNER, h, N_EMBD, D_INNER, 2, nullptr);
    }

    // final norm + tied LM head: [2048,768] x [32000,768]^T -> logits (f32)
    rmsnorm_kernel<<<SEQ_LEN, blk, 0, stream>>>(h, nfw, nfb, xn, xn_bf);
    wmma_gemm_bf16<<<dim3(VOCAB / 64, SEQ_LEN / 64), gblk, 0, stream>>>(
        xn_bf, N_EMBD, E_bf, N_EMBD, (float*)d_out, VOCAB, N_EMBD, 0, nullptr);
}
